// QuantumModel_17626545782827
// MI455X (gfx1250) — matrix operand order repaired and checked by reading, not compiled, not run on hardware
//
#include <hip/hip_runtime.h>

typedef __attribute__((ext_vector_type(16))) __bf16        v16bf;
typedef __attribute__((ext_vector_type(8)))  float         v8f;
typedef __attribute__((ext_vector_type(8)))  unsigned int  v8u;

// ---- bf16 helpers (bit-exact RNE)
static __device__ __forceinline__ unsigned short f2bfu(float f) {
  unsigned u = __builtin_bit_cast(unsigned, f);
  return (unsigned short)((u + 0x7FFFu + ((u >> 16) & 1u)) >> 16);
}
static __device__ __forceinline__ float bf2f(unsigned short h) {
  unsigned u = ((unsigned)h) << 16;
  return __builtin_bit_cast(float, u);
}

// Small zero page used as the load target for out-of-bounds conv halo pixels.
__global__ void zero_k(unsigned short* __restrict__ p, int n) {
  int i = blockIdx.x * blockDim.x + threadIdx.x;
  if (i < n) p[i] = 0;
}

// Weight repack: fp32 [O][I][kh][kw] -> bf16 [O][kh*KS+kw][I]  (K = (kh,kw,cin))
__global__ void wtrans_bf16_k(const float* __restrict__ src,
                              unsigned short* __restrict__ dst,
                              int Cout, int Cin, int KK) {
  int i = blockIdx.x * blockDim.x + threadIdx.x;
  int total = Cout * Cin * KK;
  if (i >= total) return;
  int r = i % KK; int t = i / KK; int cin = t % Cin; int cout = t / Cin;
  dst[((size_t)cout * KK + r) * Cin + cin] = f2bfu(src[i]);
}

// Input image: fp32 NCHW -> bf16 NHWC
__global__ void x_nhwc_bf16_k(const float* __restrict__ src,
                              unsigned short* __restrict__ dst,
                              int N, int C, int H, int W) {
  int i = blockIdx.x * blockDim.x + threadIdx.x;
  int total = N * C * H * W;
  if (i >= total) return;
  int w = i % W; int t = i / W; int h = t % H; t /= H; int c = t % C; int n = t / C;
  dst[(((size_t)n * H + h) * W + w) * C + c] = f2bfu(src[i]);
}

// ---------------------------------------------------------------------------
// Implicit-GEMM conv + fused eval-BN (+residual) (+ReLU); NHWC bf16 activations,
// repacked bf16 weights, f32 WMMA accumulate. One wave = 16(cout) x 32(pos)
// tile, one A fragment feeds two v_wmma_f32_16x16x32_bf16 per K-step.
// Fragment layouts used here (contiguous K order for A and B, not the ISA table order):
//   A 16x32: lane l -> M=l&15, kA=(l>>4)*16; elems = K kA..kA+15 consecutive
//            -> two b128 loads (+0, +8 elems).
//   B 32x16: lane l -> N=l&15, kB=(l>>4)*16; elems = K kB..kB+15 consecutive
//            -> two b128 loads (+0, +8 elems); NHWC makes these contiguous.
//   C/D    : VGPR r -> M = r + 8*(l>>4), N = l&15  -> 8 consecutive channels
//            per lane -> single b128 store/residual-load in NHWC.
// OOB halo handling: the per-pixel source pointer is selected ONCE per (r,t)
// (real pixel vs zero page), so the inner channel loop carries no predication.
// ---------------------------------------------------------------------------
template <int KS>
__global__ __launch_bounds__(32) void conv_bn_wmma(
    const unsigned short* __restrict__ x, const unsigned short* __restrict__ w,
    const unsigned short* __restrict__ zp,
    const float* __restrict__ bnb, const float* __restrict__ bng,
    const float* __restrict__ bnm, const float* __restrict__ bnv,
    const unsigned short* __restrict__ res, unsigned short* __restrict__ out,
    int Cin, int Cout, int Hin, int Win, int Hout, int Wout,
    int stride, int pad, int relu)
{
  constexpr int KK = KS * KS;
  const int lane  = threadIdx.x;
  const int npos  = Hout * Wout;
  const int Ktot  = Cin * KK;
  const int cout0 = blockIdx.x << 4;
  const int pos0  = blockIdx.y << 5;
  const int n     = blockIdx.z;

  const int mA = lane & 15;
  const int kA = (lane >> 4) << 4;
  const unsigned short* __restrict__ wrow = w + (size_t)(cout0 + mA) * (size_t)Ktot;

  const int kB  = (lane >> 4) << 4;
  const int col = lane & 15;
  const unsigned short* __restrict__ xin =
      x + (size_t)n * (size_t)(Hin * Win) * (size_t)Cin;

  int  ih0s[2], iw0s[2];
  bool bval[2];
#pragma unroll
  for (int t = 0; t < 2; ++t) {
    int p   = pos0 + 16 * t + col;
    bval[t] = p < npos;
    int pc  = bval[t] ? p : 0;
    int oh  = pc / Wout, ow = pc - oh * Wout;
    ih0s[t] = oh * stride - pad;
    iw0s[t] = ow * stride - pad;
  }

  v8f acc0 = {0.f, 0.f, 0.f, 0.f, 0.f, 0.f, 0.f, 0.f};
  v8f acc1 = acc0;

  if constexpr (KS != 7) {
    // Fast path: Cin % 32 == 0, so each 32-wide K-chunk sits inside one (kh,kw).
    for (int r = 0; r < KK; ++r) {
      const int kh = r / KS, kw = r - (r / KS) * KS;  // constant divisors
      const unsigned short* src[2];
#pragma unroll
      for (int t = 0; t < 2; ++t) {
        int ih = ih0s[t] + kh, iw = iw0s[t] + kw;
        bool ok = bval[t] && ((unsigned)ih < (unsigned)Hin) &&
                  ((unsigned)iw < (unsigned)Win);
        src[t] = ok ? (xin + (size_t)(ih * Win + iw) * Cin) : zp;
      }
      const unsigned short* __restrict__ wr = wrow + r * Cin;
      for (int c0 = 0; c0 < Cin; c0 += 32) {
        // A fragment: two aligned 128-bit loads
        const uint4* aq = reinterpret_cast<const uint4*>(wr + c0 + kA);
        uint4 a0 = aq[0];      // K offsets kA .. kA+7
        uint4 a1 = aq[1];      // K offsets kA+8 .. kA+15
        v8u av;
        av[0] = a0.x; av[1] = a0.y; av[2] = a0.z; av[3] = a0.w;
        av[4] = a1.x; av[5] = a1.y; av[6] = a1.z; av[7] = a1.w;
        v16bf afrag = __builtin_bit_cast(v16bf, av);

        // B fragments: 16 consecutive channels per lane -> two b128 loads,
        // no predication (OOB pixels read the zero page).
        v16bf bfr[2];
#pragma unroll
        for (int t = 0; t < 2; ++t) {
          const uint4* bq = reinterpret_cast<const uint4*>(src[t] + c0 + kB);
          uint4 b0 = bq[0];    // K kB .. kB+7
          uint4 b1 = bq[1];    // K kB+8 .. kB+15
          v8u bv;
          bv[0] = b0.x; bv[1] = b0.y; bv[2] = b0.z; bv[3] = b0.w;
          bv[4] = b1.x; bv[5] = b1.y; bv[6] = b1.z; bv[7] = b1.w;
          bfr[t] = __builtin_bit_cast(v16bf, bv);
        }
        acc0 = __builtin_amdgcn_wmma_f32_16x16x32_bf16(false, afrag, false, bfr[0],
                                                       (short)0, acc0, false, false);
        acc1 = __builtin_amdgcn_wmma_f32_16x16x32_bf16(false, afrag, false, bfr[1],
                                                       (short)0, acc1, false, false);
      }
    }
  } else {
    // Stem: Cin=3, Ktot=147 -> clamped scalar path, K = (kh*7+kw)*3 + cin.
    for (int k0 = 0; k0 < Ktot; k0 += 32) {
      v8u av;
#pragma unroll
      for (int j = 0; j < 8; ++j) {
        unsigned d = 0;
#pragma unroll
        for (int e = 0; e < 2; ++e) {
          int k  = k0 + kA + 2 * j + e;
          int kc = (k < Ktot) ? k : (Ktot - 1);
          unsigned raw = wrow[kc];
          raw = (k < Ktot) ? raw : 0u;
          d |= raw << (16 * e);
        }
        av[j] = d;
      }
      v16bf afrag = __builtin_bit_cast(v16bf, av);

      v16bf bfr[2];
#pragma unroll
      for (int t = 0; t < 2; ++t) {
        v8u bv;
#pragma unroll
        for (int j = 0; j < 8; ++j) {
          unsigned d = 0;
#pragma unroll
          for (int e = 0; e < 2; ++e) {
            int ke  = k0 + kB + 2 * j + e;
            int r   = ke / 3, cin = ke - 3 * r;
            int kh  = r / 7,  kw  = r - 7 * kh;
            int ih = ih0s[t] + kh, iw = iw0s[t] + kw;
            bool ok = bval[t] && (ke < Ktot) &&
                      ((unsigned)ih < (unsigned)Hin) &&
                      ((unsigned)iw < (unsigned)Win);
            int idx = (ih * Win + iw) * 3 + cin;
            idx = ok ? idx : 0;
            unsigned raw = xin[idx];
            raw = ok ? raw : 0u;
            d |= raw << (16 * e);
          }
          bv[j] = d;
        }
        bfr[t] = __builtin_bit_cast(v16bf, bv);
      }
      acc0 = __builtin_amdgcn_wmma_f32_16x16x32_bf16(false, afrag, false, bfr[0],
                                                     (short)0, acc0, false, false);
      acc1 = __builtin_amdgcn_wmma_f32_16x16x32_bf16(false, afrag, false, bfr[1],
                                                     (short)0, acc1, false, false);
    }
  }

  // ---- Epilogue: folded BN (+res) (+ReLU); NHWC -> one b128 store per tile
  const int rowHi = (lane >> 4) << 3;
  float scale[8], bias[8];
#pragma unroll
  for (int r = 0; r < 8; ++r) {
    int cout = cout0 + rowHi + r;
    scale[r] = bng[cout] * rsqrtf(bnv[cout] + 1e-5f);
    bias[r]  = bnb[cout] - bnm[cout] * scale[r];
  }
#pragma unroll
  for (int t = 0; t < 2; ++t) {
    int pos = pos0 + 16 * t + col;
    if (pos < npos) {
      size_t base = ((size_t)n * npos + pos) * (size_t)Cout + (cout0 + rowHi);
      float vals[8];
#pragma unroll
      for (int r = 0; r < 8; ++r)
        vals[r] = (t == 0 ? acc0[r] : acc1[r]) * scale[r] + bias[r];
      if (res) {
        uint4 rv = *reinterpret_cast<const uint4*>(res + base);
        unsigned rd[4] = {rv.x, rv.y, rv.z, rv.w};
#pragma unroll
        for (int r = 0; r < 8; ++r)
          vals[r] += bf2f((unsigned short)(rd[r >> 1] >> (16 * (r & 1))));
      }
      if (relu) {
#pragma unroll
        for (int r = 0; r < 8; ++r) vals[r] = fmaxf(vals[r], 0.f);
      }
      uint4 sv;
      sv.x = f2bfu(vals[0]) | ((unsigned)f2bfu(vals[1]) << 16);
      sv.y = f2bfu(vals[2]) | ((unsigned)f2bfu(vals[3]) << 16);
      sv.z = f2bfu(vals[4]) | ((unsigned)f2bfu(vals[5]) << 16);
      sv.w = f2bfu(vals[6]) | ((unsigned)f2bfu(vals[7]) << 16);
      *reinterpret_cast<uint4*>(out + base) = sv;
    }
  }
}

// ---------------------------------------------------------------------------
// 3x3 s2 p1 maxpool, NHWC bf16 (channel-fastest -> coalesced).
// ---------------------------------------------------------------------------
__global__ void maxpool3x3s2_nhwc_k(const unsigned short* __restrict__ in,
                                    unsigned short* __restrict__ out,
                                    int N, int C, int Hin, int Win,
                                    int Hout, int Wout)
{
  size_t total = (size_t)N * Hout * Wout * C;
  size_t i = (size_t)blockIdx.x * blockDim.x + threadIdx.x;
  if (i >= total) return;
  int c  = (int)(i % C);   size_t t = i / C;
  int ow = (int)(t % Wout); t /= Wout;
  int oh = (int)(t % Hout);
  int n  = (int)(t / Hout);
  const unsigned short* p = in + (size_t)n * Hin * Win * C;
  float m = -3.402823466e38f;
  for (int kh = 0; kh < 3; ++kh) {
    int ih = oh * 2 - 1 + kh;
    if ((unsigned)ih >= (unsigned)Hin) continue;
    for (int kw = 0; kw < 3; ++kw) {
      int iw = ow * 2 - 1 + kw;
      if ((unsigned)iw >= (unsigned)Win) continue;
      m = fmaxf(m, bf2f(p[((size_t)ih * Win + iw) * C + c]));
    }
  }
  out[i] = f2bfu(m);
}

// ---------------------------------------------------------------------------
// Head: avgpool + FC(512->4) + sigmoid*pi + exact 4-qubit statevector sim.
// One thread per batch sample; feat is NHWC [32,7,7,512] bf16.
// ---------------------------------------------------------------------------
__global__ __launch_bounds__(32) void head_quantum(
    const unsigned short* __restrict__ feat,
    const float* __restrict__ fc_w, const float* __restrict__ fc_b,
    const float* __restrict__ qw, float* __restrict__ out)
{
  int b = threadIdx.x;
  if (b >= 32) return;

  float acc[4] = {fc_b[0], fc_b[1], fc_b[2], fc_b[3]};
  const unsigned short* fb = feat + (size_t)b * 49 * 512;
  for (int c = 0; c < 512; ++c) {
    float s = 0.f;
    for (int i = 0; i < 49; ++i) s += bf2f(fb[i * 512 + c]);
    s *= (1.0f / 49.0f);
    acc[0] += fc_w[c]        * s;
    acc[1] += fc_w[512 + c]  * s;
    acc[2] += fc_w[1024 + c] * s;
    acc[3] += fc_w[1536 + c] * s;
  }
  const float PI = 3.14159265358979323846f;
  float ang[4];
  for (int j = 0; j < 4; ++j) ang[j] = PI / (1.f + expf(-acc[j]));

  float re[16], im[16];
  for (int i = 0; i < 16; ++i) { re[i] = 0.f; im[i] = 0.f; }
  re[0] = 1.f;

  for (int q = 0; q < 4; ++q) {            // RX embedding
    int mask = 8 >> q;
    float ch = cosf(0.5f * ang[q]);
    float sh = sinf(0.5f * ang[q]);
    for (int i = 0; i < 16; ++i) {
      if (i & mask) continue;
      int j = i | mask;
      float r0 = re[i], i0 = im[i], r1 = re[j], i1 = im[j];
      re[i] =  ch * r0 + sh * i1;  im[i] =  ch * i0 - sh * r1;
      re[j] =  sh * i0 + ch * r1;  im[j] = -sh * r0 + ch * i1;
    }
  }

  for (int l = 0; l < 3; ++l) {            // StronglyEntanglingLayers
    for (int q = 0; q < 4; ++q) {
      float phi = qw[(l * 4 + q) * 3 + 0];
      float th  = qw[(l * 4 + q) * 3 + 1];
      float om  = qw[(l * 4 + q) * 3 + 2];
      float ct = cosf(0.5f * th), st = sinf(0.5f * th);
      float a1 = -0.5f * (phi + om);
      float a2 =  0.5f * (phi - om);
      float c1 = cosf(a1), s1 = sinf(a1);
      float c2 = cosf(a2), s2 = sinf(a2);
      float m00r =  ct * c1, m00i =  ct * s1;
      float m01r = -st * c2, m01i = -st * s2;
      float m10r =  st * c2, m10i = -st * s2;
      float m11r =  ct * c1, m11i = -ct * s1;
      int mask = 8 >> q;
      for (int i = 0; i < 16; ++i) {
        if (i & mask) continue;
        int j = i | mask;
        float r0 = re[i], i0 = im[i], r1 = re[j], i1 = im[j];
        re[i] = m00r * r0 - m00i * i0 + m01r * r1 - m01i * i1;
        im[i] = m00r * i0 + m00i * r0 + m01r * i1 + m01i * r1;
        re[j] = m10r * r0 - m10i * i0 + m11r * r1 - m11i * i1;
        im[j] = m10r * i0 + m10i * r0 + m11r * i1 + m11i * r1;
      }
    }
    int rr = (l % 3) + 1;
    for (int q = 0; q < 4; ++q) {
      int cm = 8 >> q;
      int tm = 8 >> ((q + rr) & 3);
      for (int i = 0; i < 16; ++i) {
        if ((i & cm) && !(i & tm)) {
          int j = i | tm;
          float tr = re[i]; re[i] = re[j]; re[j] = tr;
          float ti = im[i]; im[i] = im[j]; im[j] = ti;
        }
      }
    }
  }

  float z = 0.f;
  for (int i = 0; i < 16; ++i) {
    float p = re[i] * re[i] + im[i] * im[i];
    z += (i & 8) ? -p : p;
  }
  out[b] = 0.5f * (z + 1.f);
}

// ---------------------------------------------------------------------------
// Host-side orchestration
// ---------------------------------------------------------------------------
static inline void launch_conv(hipStream_t s, const unsigned short* x,
                               const unsigned short* w, const unsigned short* zp,
                               const float* const* in, int bni,
                               const unsigned short* res, unsigned short* out,
                               int Cin, int Cout, int Hin, int Win,
                               int ks, int stride, int pad, int relu)
{
  int Hout = (Hin + 2 * pad - ks) / stride + 1;
  int Wout = (Win + 2 * pad - ks) / stride + 1;
  int npos = Hout * Wout;
  dim3 grid(Cout / 16, (npos + 31) / 32, 32);
  const float *bb = in[bni], *bg = in[bni + 1], *bm = in[bni + 2], *bv = in[bni + 3];
  if (ks == 1)
    conv_bn_wmma<1><<<grid, 32, 0, s>>>(x, w, zp, bb, bg, bm, bv, res, out,
        Cin, Cout, Hin, Win, Hout, Wout, stride, pad, relu);
  else if (ks == 3)
    conv_bn_wmma<3><<<grid, 32, 0, s>>>(x, w, zp, bb, bg, bm, bv, res, out,
        Cin, Cout, Hin, Win, Hout, Wout, stride, pad, relu);
  else
    conv_bn_wmma<7><<<grid, 32, 0, s>>>(x, w, zp, bb, bg, bm, bv, res, out,
        Cin, Cout, Hin, Win, Hout, Wout, stride, pad, relu);
}

extern "C" void kernel_launch(void* const* d_in, const int* in_sizes, int n_in,
                              void* d_out, int out_size, void* d_ws, size_t ws_size,
                              hipStream_t stream)
{
  (void)in_sizes; (void)n_in; (void)out_size; (void)ws_size;
  // Flattened-input indices assume JAX pytree order (dict keys sorted at every
  // level): params{bn1{b,g,m,v}=0..3, conv1=4, fc_b=5, fc_w=6, layers:
  //  blk = {bn1(4), bn2(4), conv1, conv2 [, down, down_bn(4)]}
  //  L0B0@7, L0B1@17, L1B0@27, L1B1@42, L2B0@52, L2B1@67, L3B0@77, L3B1@92},
  // q_weights=102, x=103.
  const float* const* in = (const float* const*)d_in;

  unsigned short* Aa = (unsigned short*)d_ws;   // bf16 arena (all NHWC)
  unsigned short* B0 = Aa;                      // 26.0M elems (stem out)
  unsigned short* B1 = Aa + 26000000ull;        //  6.5M elems
  unsigned short* B2 = Aa + 32500000ull;        //  6.5M elems
  unsigned short* XB = Aa + 39000000ull;        //  4.82M elems (bf16 NHWC image)
  unsigned short* ZP = Aa + 43900000ull;        //  1024 elems, zero page (16B aligned)
  unsigned short* WA = Aa + 44000000ull;        // 11.17M elems (repacked weights)

  zero_k<<<4, 256, 0, stream>>>(ZP, 1024);
  { int n = 32 * 3 * 224 * 224;
    x_nhwc_bf16_k<<<(n + 255) / 256, 256, 0, stream>>>(in[103], XB, 32, 3, 224, 224); }

  // conv weight tensors: index, count, Cout, Cin, KS
  const int widx[20] = {4, 15, 16, 25, 26, 35, 36, 37, 50, 51,
                        60, 61, 62, 75, 76, 85, 86, 87, 100, 101};
  const int wcnt[20] = {9408, 36864, 36864, 36864, 36864, 73728, 147456, 8192,
                        147456, 147456, 294912, 589824, 32768, 589824, 589824,
                        1179648, 2359296, 131072, 2359296, 2359296};
  const int wco[20]  = {64, 64, 64, 64, 64, 128, 128, 128, 128, 128,
                        256, 256, 256, 256, 256, 512, 512, 512, 512, 512};
  const int wci[20]  = {3, 64, 64, 64, 64, 64, 128, 64, 128, 128,
                        128, 256, 128, 256, 256, 256, 512, 256, 512, 512};
  const int wks[20]  = {7, 3, 3, 3, 3, 3, 3, 1, 3, 3,
                        3, 3, 1, 3, 3, 3, 3, 1, 3, 3};
  const unsigned short* wp[20];
  size_t off = 0;
  for (int i = 0; i < 20; ++i) {
    unsigned short* dst = WA + off;
    wtrans_bf16_k<<<(wcnt[i] + 255) / 256, 256, 0, stream>>>(
        in[widx[i]], dst, wco[i], wci[i], wks[i] * wks[i]);
    wp[i] = dst;
    off += (size_t)wcnt[i];   // all counts multiple of 8 -> 16B alignment kept
  }

  // Stem
  launch_conv(stream, XB, wp[0], ZP, in, 0, nullptr, B0, 3, 64, 224, 224, 7, 2, 3, 1);
  { int total = 32 * 64 * 56 * 56;
    maxpool3x3s2_nhwc_k<<<(total + 255) / 256, 256, 0, stream>>>(
        B0, B1, 32, 64, 112, 112, 56, 56); }
  // layer1
  launch_conv(stream, B1, wp[1], ZP, in, 7,  nullptr, B2, 64, 64, 56, 56, 3, 1, 1, 1);
  launch_conv(stream, B2, wp[2], ZP, in, 11, B1,      B0, 64, 64, 56, 56, 3, 1, 1, 1);
  launch_conv(stream, B0, wp[3], ZP, in, 17, nullptr, B2, 64, 64, 56, 56, 3, 1, 1, 1);
  launch_conv(stream, B2, wp[4], ZP, in, 21, B0,      B1, 64, 64, 56, 56, 3, 1, 1, 1);
  // layer2
  launch_conv(stream, B1, wp[5], ZP, in, 27, nullptr, B2, 64, 128, 56, 56, 3, 2, 1, 1);
  launch_conv(stream, B1, wp[7], ZP, in, 38, nullptr, B0, 64, 128, 56, 56, 1, 2, 0, 0);
  launch_conv(stream, B2, wp[6], ZP, in, 31, B0,      B1, 128, 128, 28, 28, 3, 1, 1, 1);
  launch_conv(stream, B1, wp[8], ZP, in, 42, nullptr, B2, 128, 128, 28, 28, 3, 1, 1, 1);
  launch_conv(stream, B2, wp[9], ZP, in, 46, B1,      B0, 128, 128, 28, 28, 3, 1, 1, 1);
  // layer3
  launch_conv(stream, B0, wp[10], ZP, in, 52, nullptr, B2, 128, 256, 28, 28, 3, 2, 1, 1);
  launch_conv(stream, B0, wp[12], ZP, in, 63, nullptr, B1, 128, 256, 28, 28, 1, 2, 0, 0);
  launch_conv(stream, B2, wp[11], ZP, in, 56, B1,      B0, 256, 256, 14, 14, 3, 1, 1, 1);
  launch_conv(stream, B0, wp[13], ZP, in, 67, nullptr, B2, 256, 256, 14, 14, 3, 1, 1, 1);
  launch_conv(stream, B2, wp[14], ZP, in, 71, B0,      B1, 256, 256, 14, 14, 3, 1, 1, 1);
  // layer4
  launch_conv(stream, B1, wp[15], ZP, in, 77, nullptr, B2, 256, 512, 14, 14, 3, 2, 1, 1);
  launch_conv(stream, B1, wp[17], ZP, in, 88, nullptr, B0, 256, 512, 14, 14, 1, 2, 0, 0);
  launch_conv(stream, B2, wp[16], ZP, in, 81, B0,      B1, 512, 512, 7, 7, 3, 1, 1, 1);
  launch_conv(stream, B1, wp[18], ZP, in, 92, nullptr, B2, 512, 512, 7, 7, 3, 1, 1, 1);
  launch_conv(stream, B2, wp[19], ZP, in, 96, B1,      B0, 512, 512, 7, 7, 3, 1, 1, 1);
  // head
  head_quantum<<<1, 32, 0, stream>>>(B0, in[6], in[5], in[102], (float*)d_out);
}
